// MultiHeadSelfAttention_41523743817878
// MI455X (gfx1250) — matrix operand order repaired and checked by reading, not compiled, not run on hardware
//
#include <hip/hip_runtime.h>
#include <hip/hip_bf16.h>

// ---------------------------------------------------------------------------
// MultiHeadSelfAttention for MI455X (gfx1250, wave32, WMMA)
//   B=4, T=2048, C=1024, H=16, D=64
// Stage 1: qkv = x @ W + b  (bf16 WMMA, f32 accum), double-buffered LDS
//   pipeline; emits Q (pre-scaled), K, and V^T as bf16.
// Stage 2: flash attention; K/V tiles double-buffered into LDS with
//   GLOBAL_LOAD_ASYNC_TO_LDS_B128 (ASYNCcnt), online softmax in f32,
//   S and P*V via bf16 WMMA.
// ---------------------------------------------------------------------------

#define EMB   1024
#define NH    16
#define HD    64
#define BB    4
#define TT    2048
#define BHN   (BB * NH)     // 64
#define MROWS (BB * TT)     // 8192
#define NCOLS (3 * EMB)     // 3072
#define KSTR  80            // LDS row stride (elements) for K/V/P tiles

typedef __attribute__((ext_vector_type(16))) __bf16 v16bf;
typedef __attribute__((ext_vector_type(8)))  __bf16 v8bf;
typedef __attribute__((ext_vector_type(2)))  __bf16 v2bf;
typedef __attribute__((ext_vector_type(8)))  float  v8f;
typedef __attribute__((ext_vector_type(4)))  float  v4f;
typedef __attribute__((ext_vector_type(4)))  int    v4i;
typedef __attribute__((ext_vector_type(2)))  unsigned v2u;

typedef __attribute__((address_space(1))) v4i g_v4i;   // global int4
typedef __attribute__((address_space(3))) v4i l_v4i;   // LDS int4
typedef __attribute__((address_space(3))) __bf16 lds_bf16;

__device__ __forceinline__ __bf16 f2bf(float f) {
  unsigned u = __builtin_bit_cast(unsigned, f);
  u += 0x7fffu + ((u >> 16) & 1u);          // round-to-nearest-even
  unsigned short s = (unsigned short)(u >> 16);
  return __builtin_bit_cast(__bf16, s);
}

// Pack two floats into one dword of bf16 (low = a, high = b).
__device__ __forceinline__ unsigned cvt2bf(float a, float b) {
#if __has_builtin(__builtin_amdgcn_cvt_pk_bf16_f32)
  v2bf pk = __builtin_amdgcn_cvt_pk_bf16_f32(a, b);
  return __builtin_bit_cast(unsigned, pk);
#else
  return (unsigned)__builtin_bit_cast(unsigned short, f2bf(a)) |
         ((unsigned)__builtin_bit_cast(unsigned short, f2bf(b)) << 16);
#endif
}

__device__ __forceinline__ v16bf cat16(v8bf lo, v8bf hi) {
  union { v16bf w; v8bf h[2]; } u;
  u.h[0] = lo; u.h[1] = hi;
  return u.w;
}

__device__ __forceinline__ v8f zero8() {
  v8f z;
  #pragma unroll
  for (int i = 0; i < 8; ++i) z[i] = 0.0f;
  return z;
}

// Async 16-byte global->LDS copy (tracked by ASYNCcnt, no VGPR round trip).
__device__ __forceinline__ void async_b128(__bf16* lds_dst, const __bf16* gsrc) {
#if __has_builtin(__builtin_amdgcn_global_load_async_to_lds_b128)
  __builtin_amdgcn_global_load_async_to_lds_b128(
      (g_v4i*)gsrc, (l_v4i*)lds_dst, /*offset=*/0, /*cpol=*/0);
#else
  asm volatile("global_load_async_to_lds_b128 %0, %1, off"
               : : "v"((unsigned)(size_t)(lds_bf16*)lds_dst), "v"(gsrc)
               : "memory");
#endif
}

__device__ __forceinline__ void wait_async0() {
#if __has_builtin(__builtin_amdgcn_s_wait_asynccnt)
  __builtin_amdgcn_s_wait_asynccnt(0);
#else
  asm volatile("s_wait_asynccnt 0x0" : : : "memory");
#endif
}

// A-fragment, 16x32 tile of a row-major [rows][ld] bf16 matrix.
// ISA 7.12.2 (16-bit A 16x32): lane L holds row (L&15); VGPR0-3 = K {0..7}
// (lanes<16) / {8..15} (lanes>=16); VGPR4-7 = +16.
__device__ __forceinline__ v16bf load_frag_a(const __bf16* p, int rbase,
                                             int kbase, int ld, int lane) {
  const int row = rbase + (lane & 15);
  const int k0  = kbase + ((lane & 16) ? 8 : 0);
  const __bf16* b = p + row * ld + k0;
  v8bf lo = *(const v8bf*)(b);
  v8bf hi = *(const v8bf*)(b + 16);
  return cat16(lo, hi);
}

// B-fragment, 32x16 tile whose memory layout is [n][k] row-major (K-dim
// contiguous per output column).  Lane L holds column (L&15); K {0..7}
// (lanes<16) / {8..15} (lanes>=16), then +16, matching the A-fragment.
__device__ __forceinline__ v16bf load_frag_b(const __bf16* p, int nbase,
                                             int kbase, int ld, int lane) {
  const int n  = nbase + (lane & 15);
  const int k0 = kbase + ((lane & 16) ? 8 : 0);
  const __bf16* b = p + n * ld + k0;
  v8bf lo = *(const v8bf*)(b);
  v8bf hi = *(const v8bf*)(b + 16);
  return cat16(lo, hi);
}

__device__ __forceinline__ v8f bwmma(v16bf a, v16bf b, v8f c) {
  return __builtin_amdgcn_wmma_f32_16x16x32_bf16(
      /*neg_a=*/false, a, /*neg_b=*/false, b,
      /*c_mod=*/(short)0, c, /*reuse_a=*/false, /*reuse_b=*/false);
}

// ---------------------------------------------------------------------------
// Kernel 1: QKV projection GEMM.  Block tile 128x128, BK=32, 8 waves
// (4x2 wave grid, each wave 32x64 = 2x4 WMMA tiles).  LDS double-buffered:
// global loads for step i+1 overlap the WMMAs of step i.
// ---------------------------------------------------------------------------
__global__ __launch_bounds__(256)
void mhsa_qkv_gemm(const float* __restrict__ x, const float* __restrict__ W,
                   const float* __restrict__ bias,
                   __bf16* __restrict__ qb, __bf16* __restrict__ kb,
                   __bf16* __restrict__ vb) {
  __shared__ __bf16 As[2][128 * 48];   // [m][k], stride 48 (16B-aligned rows)
  __shared__ __bf16 Bs[2][128 * 48];   // [n][k] (transposed on store)

  const int tid  = threadIdx.x;
  const int lane = tid & 31;
  const int wav  = tid >> 5;
  const int n0   = blockIdx.x * 128;
  const int m0   = blockIdx.y * 128;
  const int region = n0 >> 10;      // 0=Q 1=K 2=V, uniform per block

  const int wm = wav >> 1;          // 0..3
  const int wn = wav & 1;           // 0..1

  v8f acc[2][4];
  #pragma unroll
  for (int mi = 0; mi < 2; ++mi)
    #pragma unroll
    for (int ni = 0; ni < 4; ++ni) acc[mi][ni] = zero8();

  const int arow = tid >> 3;            // 0..31
  const int acol = (tid & 7) * 4;       // 0,4,..28
  const int brow = tid >> 3;            // k row 0..31
  const int bcol = (tid & 7) * 16;      // n col 0,16,..112

  v4f va[4], wb4[4];
  auto load_regs = [&](int k0) {
    #pragma unroll
    for (int p = 0; p < 4; ++p)
      va[p] = *(const v4f*)(x + (size_t)(m0 + arow + p * 32) * EMB + k0 + acol);
    const float* src = W + (size_t)(k0 + brow) * NCOLS + n0 + bcol;
    #pragma unroll
    for (int q4 = 0; q4 < 4; ++q4)
      wb4[q4] = *(const v4f*)(src + q4 * 4);
  };
  auto store_tile = [&](int buf) {
    #pragma unroll
    for (int p = 0; p < 4; ++p) {
      v2u pk;
      pk.x = cvt2bf(va[p].x, va[p].y);
      pk.y = cvt2bf(va[p].z, va[p].w);
      *(v2u*)&As[buf][(arow + p * 32) * 48 + acol] = pk;
    }
    #pragma unroll
    for (int q4 = 0; q4 < 4; ++q4) {
      const int cb = bcol + q4 * 4;
      Bs[buf][(cb + 0) * 48 + brow] = f2bf(wb4[q4].x);
      Bs[buf][(cb + 1) * 48 + brow] = f2bf(wb4[q4].y);
      Bs[buf][(cb + 2) * 48 + brow] = f2bf(wb4[q4].z);
      Bs[buf][(cb + 3) * 48 + brow] = f2bf(wb4[q4].w);
    }
  };

  // Prologue: stage k-step 0.
  load_regs(0);
  store_tile(0);
  __syncthreads();

  for (int i = 0; i < EMB / 32; ++i) {
    const int cur = i & 1;
    if (i + 1 < EMB / 32) load_regs((i + 1) * 32);   // overlap with WMMA

    v16bf af[2], bf[4];
    #pragma unroll
    for (int mi = 0; mi < 2; ++mi)
      af[mi] = load_frag_a(&As[cur][0], wm * 32 + mi * 16, 0, 48, lane);
    #pragma unroll
    for (int ni = 0; ni < 4; ++ni)
      bf[ni] = load_frag_b(&Bs[cur][0], wn * 64 + ni * 16, 0, 48, lane);
    #pragma unroll
    for (int mi = 0; mi < 2; ++mi)
      #pragma unroll
      for (int ni = 0; ni < 4; ++ni)
        acc[mi][ni] = bwmma(af[mi], bf[ni], acc[mi][ni]);

    if (i + 1 < EMB / 32) store_tile(cur ^ 1);
    __syncthreads();
  }

  // Epilogue: bias add, scatter into Q / K / V^T bf16 buffers.
  const float scl = 0.03125f;                 // 1/sqrt(1024)
  const int roff  = (lane & 16) ? 8 : 0;
  #pragma unroll
  for (int mi = 0; mi < 2; ++mi) {
    #pragma unroll
    for (int ni = 0; ni < 4; ++ni) {
      const int ncol = n0 + wn * 64 + ni * 16 + (lane & 15);
      const float bv = bias[ncol];
      const int c = ncol & (EMB - 1);
      const int h = c >> 6, d = c & 63;
      #pragma unroll
      for (int r = 0; r < 8; ++r) {
        const int m = m0 + wm * 32 + mi * 16 + r + roff;
        const int b = m >> 11, t = m & (TT - 1);
        const int bh = b * NH + h;
        const float val = acc[mi][ni][r] + bv;
        if (region == 0) {
          qb[((size_t)bh * TT + t) * HD + d] = f2bf(val * scl);
        } else if (region == 1) {
          kb[((size_t)bh * TT + t) * HD + d] = f2bf(val);
        } else {
          vb[((size_t)bh * HD + d) * TT + t] = f2bf(val);   // V transposed
        }
      }
    }
  }
}

// ---------------------------------------------------------------------------
// Kernel 2: causal flash attention.  One block per (64-query tile, b*h);
// 4 waves, each owns 16 query rows.  All waves share the same key blocks,
// so K (64x64) and V^T (64x64) tiles are staged once per block into LDS
// with async global->LDS copies, double-buffered across key blocks.
// ---------------------------------------------------------------------------
__global__ __launch_bounds__(128)
void mhsa_flash_attn(const __bf16* __restrict__ qb,
                     const __bf16* __restrict__ kb,
                     const __bf16* __restrict__ vt,
                     float* __restrict__ out) {
  __shared__ __bf16 Ks[2][64 * KSTR];   // [key][d]   10 KB per buffer
  __shared__ __bf16 Vs[2][64 * KSTR];   // [d][key]   10 KB per buffer
  __shared__ __bf16 Pbuf[4][16 * KSTR]; // per-wave P transpose buffer

  const int tid   = threadIdx.x;
  const int lane  = tid & 31;
  const int wav   = tid >> 5;
  const int qt    = blockIdx.x;         // 0..31
  const int bh    = blockIdx.y;         // 0..63
  const int qbase = qt * 64 + wav * 16;
  const int roff  = (lane & 16) ? 8 : 0;
  const int ncol  = lane & 15;

  const __bf16* Q = qb + (size_t)bh * TT * HD;
  const __bf16* K = kb + (size_t)bh * TT * HD;
  const __bf16* V = vt + (size_t)bh * HD * TT;

  // Cooperative async stage of one 64-key block: K rows + V^T rows.
  // 128 threads x 16B x 4 passes = 8 KB per matrix.
  const int srow = tid >> 3;            // 0..15
  const int scol = (tid & 7) * 8;       // element offset, 16B granules
  auto stage = [&](int buf, int kbase) {
    #pragma unroll
    for (int p = 0; p < 4; ++p) {
      const int row = srow + p * 16;    // key row for K, d row for V^T
      async_b128(&Ks[buf][row * KSTR + scol],
                 K + (size_t)(kbase + row) * HD + scol);
      async_b128(&Vs[buf][row * KSTR + scol],
                 V + (size_t)row * TT + kbase + scol);
    }
  };

  // Q strip (16x64) held in two A fragments for the whole kernel.
  const v16bf aq0 = load_frag_a(Q, qbase, 0,  HD, lane);
  const v16bf aq1 = load_frag_a(Q, qbase, 32, HD, lane);

  v8f o[4];
  #pragma unroll
  for (int dt = 0; dt < 4; ++dt) o[dt] = zero8();
  float mrow[8], lrow[8];
  #pragma unroll
  for (int r = 0; r < 8; ++r) { mrow[r] = -__builtin_inff(); lrow[r] = 0.0f; }

  stage(0, 0);                          // prologue: key block 0 in flight

  for (int j = 0; j <= qt; ++j) {
    const int cur = j & 1;
    const int kbase = j * 64;
    const bool diag = (j == qt);

    wait_async0();                      // own copies for block j done
    __syncthreads();                    // everyone's copies for block j done

    if (j < qt) stage(cur ^ 1, kbase + 64);   // overlap next block's copy

    // S = Q * K^T  (16 x 64 strip, 4 column tiles) from LDS
    v8f s[4];
    #pragma unroll
    for (int nt = 0; nt < 4; ++nt) {
      v16bf bk0 = load_frag_b(&Ks[cur][0], nt * 16, 0,  KSTR, lane);
      v16bf bk1 = load_frag_b(&Ks[cur][0], nt * 16, 32, KSTR, lane);
      v8f acc = zero8();
      acc = bwmma(aq0, bk0, acc);
      acc = bwmma(aq1, bk1, acc);
      s[nt] = acc;
    }

    if (diag) {
      #pragma unroll
      for (int nt = 0; nt < 4; ++nt)
        #pragma unroll
        for (int r = 0; r < 8; ++r) {
          const int kg = kbase + nt * 16 + ncol;
          const int qg = qbase + r + roff;
          if (kg > qg) s[nt][r] = -__builtin_inff();
        }
    }

    // Online softmax: row max across the 16 lanes of each half-wave.
    float alpha[8];
    #pragma unroll
    for (int r = 0; r < 8; ++r) {
      float mx = fmaxf(fmaxf(s[0][r], s[1][r]), fmaxf(s[2][r], s[3][r]));
      #pragma unroll
      for (int off = 1; off < 16; off <<= 1)
        mx = fmaxf(mx, __shfl_xor(mx, off, 32));
      const float mnew = fmaxf(mrow[r], mx);
      alpha[r] = __expf(mrow[r] - mnew);
      mrow[r]  = mnew;
    }

    // P = exp(S - m); stash bf16 P into per-wave LDS for layout change.
    #pragma unroll
    for (int nt = 0; nt < 4; ++nt)
      #pragma unroll
      for (int r = 0; r < 8; ++r) {
        const float p = __expf(s[nt][r] - mrow[r]);
        s[nt][r] = p;
        Pbuf[wav][(r + roff) * KSTR + nt * 16 + ncol] = f2bf(p);
      }

    #pragma unroll
    for (int r = 0; r < 8; ++r) {
      float sum = (s[0][r] + s[1][r]) + (s[2][r] + s[3][r]);
      #pragma unroll
      for (int off = 1; off < 16; off <<= 1)
        sum += __shfl_xor(sum, off, 32);
      lrow[r] = lrow[r] * alpha[r] + sum;
      #pragma unroll
      for (int dt = 0; dt < 4; ++dt) o[dt][r] *= alpha[r];
    }

    // Same-wave LDS RAW: DS ops are in-order per wave; just stop the
    // compiler from reordering the loads above the stores.
    asm volatile("" ::: "memory");

    // O += P * V   (V tile in LDS as [d][key], K-dim contiguous)
    #pragma unroll
    for (int cchunk = 0; cchunk < 2; ++cchunk) {
      v16bf ap = load_frag_a(&Pbuf[wav][0], 0, cchunk * 32, KSTR, lane);
      #pragma unroll
      for (int dt = 0; dt < 4; ++dt) {
        v16bf bv = load_frag_b(&Vs[cur][0], dt * 16, cchunk * 32, KSTR, lane);
        o[dt] = bwmma(ap, bv, o[dt]);
      }
    }
  }

  // Epilogue: O / l, write f32 out[bh][t][d].
  #pragma unroll
  for (int r = 0; r < 8; ++r) {
    const float inv = 1.0f / lrow[r];
    const int q = qbase + r + roff;
    #pragma unroll
    for (int dt = 0; dt < 4; ++dt)
      out[((size_t)bh * TT + q) * HD + dt * 16 + ncol] = o[dt][r] * inv;
  }
}

// ---------------------------------------------------------------------------
extern "C" void kernel_launch(void* const* d_in, const int* in_sizes, int n_in,
                              void* d_out, int out_size, void* d_ws,
                              size_t ws_size, hipStream_t stream) {
  const float* x    = (const float*)d_in[0];
  const float* W    = (const float*)d_in[1];
  const float* bias = (const float*)d_in[2];
  float* out        = (float*)d_out;

  __bf16* ws = (__bf16*)d_ws;
  const size_t qkv_elems = (size_t)BHN * TT * HD;   // 8388608
  __bf16* qb = ws;
  __bf16* kb = ws + qkv_elems;
  __bf16* vb = ws + 2 * qkv_elems;

  dim3 g1(NCOLS / 128, MROWS / 128);   // 24 x 64
  mhsa_qkv_gemm<<<g1, dim3(256), 0, stream>>>(x, W, bias, qb, kb, vb);

  dim3 g2(TT / 64, BHN);               // 32 x 64
  mhsa_flash_attn<<<g2, dim3(128), 0, stream>>>(qb, kb, vb, out);
}
